// MHA_30477087932909
// MI455X (gfx1250) — compile-verified, matrix operand order checked by reading, not run on hardware
//
#include <hip/hip_runtime.h>
#include <stdint.h>

typedef __attribute__((ext_vector_type(16))) __bf16 v16bf;
typedef __attribute__((ext_vector_type(8)))  float  v8f;

#define S_LEN 4096
#define DM    1024
#define NH    32
#define CC    3072      // C_CONV = D_MODEL + 2*D_QK
#define KCONV 4
#define CHUNK 128
#define NC    (S_LEN / CHUNK)   // 32 chunks per head

static __device__ __forceinline__ unsigned short f32_to_bf16(float f) {
  unsigned int u = __float_as_uint(f);
  u += 0x7fffu + ((u >> 16) & 1u);          // round to nearest even
  return (unsigned short)(u >> 16);
}
static __device__ __forceinline__ float bf16_to_f32(unsigned short h) {
  return __uint_as_float(((unsigned int)h) << 16);
}

// ---------------------------------------------------------------- convert
__global__ __launch_bounds__(256) void cvt_f32_to_bf16(
    const float* __restrict__ src, unsigned short* __restrict__ dst, int n) {
  int i = blockIdx.x * blockDim.x + threadIdx.x;
  int stride = gridDim.x * blockDim.x;
  for (; i < n; i += stride) dst[i] = f32_to_bf16(src[i]);
}

// ---------------------------------------------------------------- GEMM (NT)
// C[m,n] = sum_k A[m,k]*B[n,k] + bias[n]
// A: MxK bf16 row-major, B: NxK bf16 row-major (both K-contiguous).
// Wave tile: (WTM*16) x (WTN*16).  Wave grid in block: BWM x BWN.
// Block tile: (BWM*WTM*16) x (BWN*WTN*16), blockDim = BWM*BWN*32.
template <int WTM, int WTN, int BWM, int BWN, int STORE_BF16>
__global__ __launch_bounds__(BWM * BWN * 32) void gemm_bf16_nt(
    const unsigned short* __restrict__ A, const unsigned short* __restrict__ B,
    const float* __restrict__ bias, void* __restrict__ Cout, int N, int K)
{
  const int lane = threadIdx.x & 31;
  const int wave = threadIdx.x >> 5;
  const int wm = wave % BWM;
  const int wn = wave / BWM;
  const int m0 = blockIdx.y * (BWM * WTM * 16) + wm * (WTM * 16);
  const int n0 = blockIdx.x * (BWN * WTN * 16) + wn * (WTN * 16);

  const int lrow = lane & 15;
  const int koff = (lane >> 4) << 3;       // 0 for lanes 0-15, 8 for lanes 16-31

  v8f acc[WTM][WTN] = {};

  const unsigned short* Ar[WTM];
  const unsigned short* Br[WTN];
  #pragma unroll
  for (int i = 0; i < WTM; ++i) Ar[i] = A + (size_t)(m0 + i * 16 + lrow) * K + koff;
  #pragma unroll
  for (int j = 0; j < WTN; ++j) Br[j] = B + (size_t)(n0 + j * 16 + lrow) * K + koff;

  union Frag { uint4 u[2]; v16bf v; };

  #pragma unroll 1
  for (int k = 0; k < K; k += 32) {
    Frag af[WTM], bfr[WTN];
    #pragma unroll
    for (int i = 0; i < WTM; ++i) {
      af[i].u[0] = *(const uint4*)(Ar[i] + k);
      af[i].u[1] = *(const uint4*)(Ar[i] + k + 16);
    }
    #pragma unroll
    for (int j = 0; j < WTN; ++j) {
      bfr[j].u[0] = *(const uint4*)(Br[j] + k);
      bfr[j].u[1] = *(const uint4*)(Br[j] + k + 16);
    }
    #pragma unroll
    for (int i = 0; i < WTM; ++i)
      #pragma unroll
      for (int j = 0; j < WTN; ++j)
        acc[i][j] = __builtin_amdgcn_wmma_f32_16x16x32_bf16(
            false, af[i].v, false, bfr[j].v, (short)0, acc[i][j], false, false);
  }

  // C layout (f32 16x16): VGPR r -> M = r (lanes 0-15) / r+8 (lanes 16-31), N = lane&15
  #pragma unroll
  for (int i = 0; i < WTM; ++i) {
    const int mt = m0 + i * 16 + ((lane >> 4) << 3);
    #pragma unroll
    for (int j = 0; j < WTN; ++j) {
      const int nt = n0 + j * 16 + (lane & 15);
      const float bn = bias[nt];
      if (STORE_BF16) {
        unsigned short* C = (unsigned short*)Cout;
        #pragma unroll
        for (int r = 0; r < 8; ++r)
          C[(size_t)(mt + r) * N + nt] = f32_to_bf16(acc[i][j][r] + bn);
      } else {
        float* C = (float*)Cout;
        #pragma unroll
        for (int r = 0; r < 8; ++r)
          C[(size_t)(mt + r) * N + nt] = acc[i][j][r] + bn;
      }
    }
  }
}

// -------------------------------------------- depthwise causal conv + SiLU + split
// qkvb: [S, CC] bf16. Outputs q/k (elu+1) and v, layout [H][S][32] fp32.
__global__ __launch_bounds__(256) void conv_silu_split(
    const unsigned short* __restrict__ qkvb,
    const float* __restrict__ conv_w, const float* __restrict__ conv_b,
    float* __restrict__ qf, float* __restrict__ kf, float* __restrict__ vf)
{
  int idx = blockIdx.x * blockDim.x + threadIdx.x;
  if (idx >= S_LEN * CC) return;
  const int s = idx / CC;
  const int c = idx % CC;

  float acc = conv_b[c];
  #pragma unroll
  for (int t = 0; t < KCONV; ++t) {
    const int ss = s - (KCONV - 1) + t;
    if (ss >= 0)
      acc += bf16_to_f32(qkvb[(size_t)ss * CC + c]) * conv_w[c * KCONV + t];
  }
  const float z = acc / (1.f + __expf(-acc));   // SiLU

  const int grp = c >> 10;        // 0:q 1:k 2:v
  const int cc  = c & 1023;
  const int h = cc >> 5, d = cc & 31;
  const size_t o = ((size_t)h * S_LEN + s) * 32 + d;
  if (grp == 0)      qf[o] = (z > 0.f) ? z + 1.f : __expf(z);   // elu+1
  else if (grp == 1) kf[o] = (z > 0.f) ? z + 1.f : __expf(z);
  else               vf[o] = z;
}

// ---------------------------------------------------------- pass 1: chunk sums
// block = 1024 threads (32 waves): wave = t (v dim), lane = d (k dim)
__global__ __launch_bounds__(1024) void chunk_sums(
    const float* __restrict__ kf, const float* __restrict__ vf,
    float* __restrict__ ckv, float* __restrict__ cks)
{
  const int c = blockIdx.x, h = blockIdx.y;
  const int t = threadIdx.x >> 5, d = threadIdx.x & 31;
  const size_t base = ((size_t)h * S_LEN + (size_t)c * CHUNK) * 32;

  float kvs = 0.f, kc = 0.f;
  for (int s = 0; s < CHUNK; ++s) {
    const float kd = kf[base + s * 32 + d];
    const float vt = vf[base + s * 32 + t];
    kvs = fmaf(kd, vt, kvs);
    kc += kd;
  }
  ckv[(((size_t)h * NC + c) * 32 + t) * 32 + d] = kvs;
  if (t == 0) cks[((size_t)h * NC + c) * 32 + d] = kc;
}

// --------------------------------------------- pass 2: exclusive prefix over chunks
__global__ __launch_bounds__(1024) void scan_chunks(
    float* __restrict__ ckv, float* __restrict__ cks)
{
  const int h = blockIdx.x;
  const int t = threadIdx.x >> 5, d = threadIdx.x & 31;
  float carry = 0.f;
  for (int c = 0; c < NC; ++c) {
    const size_t idx = (((size_t)h * NC + c) * 32 + t) * 32 + d;
    const float v = ckv[idx];
    ckv[idx] = carry;
    carry += v;
  }
  if (threadIdx.x < 32) {
    float carry2 = 0.f;
    for (int c = 0; c < NC; ++c) {
      const size_t idx = ((size_t)h * NC + c) * 32 + d;
      const float v = cks[idx];
      cks[idx] = carry2;
      carry2 += v;
    }
  }
}

// ---------------------------------- pass 3: within-chunk inclusive scan + context
__global__ __launch_bounds__(1024) void chunk_scan_out(
    const float* __restrict__ qf, const float* __restrict__ kf,
    const float* __restrict__ vf,
    const float* __restrict__ ckv, const float* __restrict__ cks,
    unsigned short* __restrict__ ctxb)
{
  const int c = blockIdx.x, h = blockIdx.y;
  const int t = threadIdx.x >> 5, d = threadIdx.x & 31;
  const size_t base = ((size_t)h * S_LEN + (size_t)c * CHUNK) * 32;

  float kvs = ckv[(((size_t)h * NC + c) * 32 + t) * 32 + d];
  float kc  = cks[((size_t)h * NC + c) * 32 + d];

  for (int s = 0; s < CHUNK; ++s) {
    const float kd = kf[base + s * 32 + d];
    const float vt = vf[base + s * 32 + t];
    const float qd = qf[base + s * 32 + d];
    kvs = fmaf(kd, vt, kvs);
    kc += kd;
    float num = qd * kvs;
    float den = qd * kc;
    #pragma unroll
    for (int off = 16; off > 0; off >>= 1) {   // wave32 butterfly reduce over d
      num += __shfl_xor(num, off, 32);
      den += __shfl_xor(den, off, 32);
    }
    if (d == 0) {
      const int sg = c * CHUNK + s;
      ctxb[(size_t)sg * DM + h * 32 + t] = f32_to_bf16(num / den);
    }
  }
}

// ---------------------------------------------------------------- launcher
extern "C" void kernel_launch(void* const* d_in, const int* in_sizes, int n_in,
                              void* d_out, int out_size, void* d_ws, size_t ws_size,
                              hipStream_t stream) {
  const float* x      = (const float*)d_in[0];
  const float* Wqkv_w = (const float*)d_in[1];
  const float* Wqkv_b = (const float*)d_in[2];
  const float* conv_w = (const float*)d_in[3];
  const float* conv_b = (const float*)d_in[4];
  const float* out_w  = (const float*)d_in[5];
  const float* out_b  = (const float*)d_in[6];
  (void)in_sizes; (void)n_in; (void)out_size; (void)ws_size;

  char* ws = (char*)d_ws;
  size_t off = 0;
  auto carve = [&](size_t bytes) -> void* {
    void* p = ws + off;
    off = (off + bytes + 255) & ~(size_t)255;
    return p;
  };
  unsigned short* xb   = (unsigned short*)carve((size_t)S_LEN * DM * 2);   //  8 MB
  unsigned short* wqkb = (unsigned short*)carve((size_t)CC * DM * 2);      //  6 MB
  unsigned short* owb  = (unsigned short*)carve((size_t)DM * DM * 2);      //  2 MB
  unsigned short* qkvb = (unsigned short*)carve((size_t)S_LEN * CC * 2);   // 24 MB
  float* qf = (float*)carve((size_t)S_LEN * DM * 4);                       // 16 MB
  float* kf = (float*)carve((size_t)S_LEN * DM * 4);                       // 16 MB
  float* vf = (float*)carve((size_t)S_LEN * DM * 4);                       // 16 MB
  unsigned short* ctxb = (unsigned short*)carve((size_t)S_LEN * DM * 2);   //  8 MB
  float* ckv = (float*)carve((size_t)NH * NC * 32 * 32 * 4);               //  4 MB
  float* cks = (float*)carve((size_t)NH * NC * 32 * 4);                    // 128 KB

  // 1) fp32 -> bf16 operand conversion
  {
    int n = S_LEN * DM;
    cvt_f32_to_bf16<<<(n + 255) / 256, 256, 0, stream>>>(x, xb, n);
    n = CC * DM;
    cvt_f32_to_bf16<<<(n + 255) / 256, 256, 0, stream>>>(Wqkv_w, wqkb, n);
    n = DM * DM;
    cvt_f32_to_bf16<<<(n + 255) / 256, 256, 0, stream>>>(out_w, owb, n);
  }

  // 2) QKV projection: [4096,1024] x [3072,1024]^T -> [4096,3072] (bf16 out)
  //    64x64 wave tiles, 128x256 block tile -> 384 blocks, 32 FLOP/L2-byte
  gemm_bf16_nt<4, 4, 2, 4, 1><<<dim3(CC / 256, S_LEN / 128), 256, 0, stream>>>(
      xb, wqkb, Wqkv_b, qkvb, CC, DM);

  // 3) depthwise causal conv(K=4) + bias + SiLU, split into q/k (elu+1) and v
  {
    int n = S_LEN * CC;
    conv_silu_split<<<(n + 255) / 256, 256, 0, stream>>>(
        qkvb, conv_w, conv_b, qf, kf, vf);
  }

  // 4) chunked linear-attention scan
  chunk_sums<<<dim3(NC, NH), 1024, 0, stream>>>(kf, vf, ckv, cks);
  scan_chunks<<<NH, 1024, 0, stream>>>(ckv, cks);
  chunk_scan_out<<<dim3(NC, NH), 1024, 0, stream>>>(qf, kf, vf, ckv, cks, ctxb);

  // 5) output projection: ctx [4096,1024] x out_w[1024,1024]^T + out_b -> f32 out
  //    32x64 wave tiles, 128x128 block tile -> 256 blocks / 2048 waves
  gemm_bf16_nt<2, 4, 4, 2, 0><<<dim3(DM / 128, S_LEN / 128), 256, 0, stream>>>(
      ctxb, owb, out_b, d_out, DM, DM);
}
